// LSTMModel_618475291218
// MI455X (gfx1250) — matrix operand order repaired and checked by reading, not compiled, not run on hardware
//
#include <hip/hip_runtime.h>
#include <hip/hip_bf16.h>

// ---------------------------------------------------------------------------
// GCLSTM (H0=C0=0 => Cheb terms reduce to biases) + GCNConv + BN + Linear
// Dense GEMMs via V_WMMA_F32_16X16X32_BF16 (CDNA5 native matrix path).
// ---------------------------------------------------------------------------

typedef __attribute__((ext_vector_type(16))) __bf16 v16bf;
typedef __attribute__((ext_vector_type(8)))  __bf16 v8bf;
typedef __attribute__((ext_vector_type(8)))  float  v8f;

constexpr int Nn   = 50000;
constexpr int Ee   = 640000;
constexpr int Hc   = 128;
constexpr int NPAD = 50048;           // 391 * 128 (pad rows for full WMMA tiles)

// workspace layout (bytes, all offsets 256-aligned)
constexpr size_t OFF_SWZ  = 0;                                  // 4*128*128 bf16 (swizzled weights)
constexpr size_t OFF_BIAS = 131072;                             // 3*128 f32 fused gate biases
constexpr size_t OFF_DEG  = OFF_BIAS + 2048;                    // NPAD f32
constexpr size_t OFF_DG   = OFF_DEG  + (size_t)NPAD * 4;        // NPAD f32
constexpr size_t OFF_HS   = OFF_DG   + (size_t)NPAD * 4;        // NPAD*128 bf16
constexpr size_t OFF_H    = OFF_HS   + (size_t)NPAD * Hc * 2;   // NPAD*128 f32
constexpr size_t OFF_G    = OFF_H    + (size_t)NPAD * Hc * 4;   // NPAD*128 f32

__device__ __forceinline__ float sigm(float x) { return 1.0f / (1.0f + __expf(-x)); }

__device__ __forceinline__ v8f wmma_bf16(v16bf a, v16bf b, v8f c) {
  // D = A(16x32 bf16) x B(32x16 bf16) + C(16x16 f32)
  return __builtin_amdgcn_wmma_f32_16x16x32_bf16(false, a, false, b, (short)0, c,
                                                 false, false);
}

// ---------------------------------------------------------------------------
// Prep: convert live weight matrices {Wx[0], Wx[2], Wx[3], gcn_w} to bf16 in
// WMMA B-fragment order:  swz[((mat*8+nt)*4+kt)*32 + lane][j]
//   lane: n = nt*16 + (lane&15), half = lane>>4 ; element j<8 -> K = kt*32 + half*8 + j, j>=8 -> K = kt*32 + 16 + half*8 + (j-8)
// Also fuse gate biases: bias = theta_b + b for gates i,c,o.
// ---------------------------------------------------------------------------
__global__ void prep_kernel(const float* __restrict__ Wx, const float* __restrict__ gcnw,
                            const float* __restrict__ theta_b, const float* __restrict__ b,
                            __bf16* __restrict__ swz, float* __restrict__ bias) {
  int o = blockIdx.x * blockDim.x + threadIdx.x;      // 65536 total
  int j   = o & 15;
  int L   = (o >> 4)  & 31;
  int kt  = (o >> 9)  & 3;
  int nt  = (o >> 11) & 7;
  int mat = (o >> 14) & 3;
  int n = nt * 16 + (L & 15);
  int k = kt * 32 + (L >> 4) * 8 + (j & 7) + (j >> 3) * 16;
  const int gate_map[3] = {0, 2, 3};                  // i, c, o
  float v;
  if (mat < 3) v = Wx[(size_t)gate_map[mat] * Hc * Hc + (size_t)k * Hc + n];
  else         v = gcnw[(size_t)k * Hc + n];
  swz[o] = (__bf16)v;

  if (blockIdx.x == 0 && threadIdx.x < (unsigned)Hc) {
    int t = threadIdx.x;
    bias[t]          = theta_b[0 * Hc + t] + b[0 * Hc + t];   // gate i
    bias[Hc + t]     = theta_b[2 * Hc + t] + b[2 * Hc + t];   // gate c
    bias[2 * Hc + t] = theta_b[3 * Hc + t] + b[3 * Hc + t];   // gate o
  }
}

__global__ void deg_init_kernel(float* __restrict__ deg) {
  int i = blockIdx.x * blockDim.x + threadIdx.x;
  if (i < NPAD) deg[i] = 1.0f;                        // self loop
}

__global__ void deg_count_kernel(const int* __restrict__ ei, float* __restrict__ deg) {
  int e = blockIdx.x * blockDim.x + threadIdx.x;
  if (e < Ee) atomicAdd(&deg[ei[Ee + e]], 1.0f);      // dst = edge_index[1]
}

__global__ void dg_kernel(const float* __restrict__ deg, float* __restrict__ dg) {
  int i = blockIdx.x * blockDim.x + threadIdx.x;
  if (i < NPAD) dg[i] = rsqrtf(deg[i]);
}

// ---------------------------------------------------------------------------
// Gates: Hs = O * tanh(I*T) fused on top of 3 WMMA GEMMs (x @ Wx[i,c,o]).
// Block = 256 threads = 8 waves; block handles 128 rows; wave handles 16 rows.
// ---------------------------------------------------------------------------
__global__ __launch_bounds__(256)
void gates_wmma_kernel(const float* __restrict__ x, const __bf16* __restrict__ swz,
                       const float* __restrict__ bias, const float* __restrict__ w_c,
                       __bf16* __restrict__ Hs) {
  __shared__ __bf16 xt[128 * Hc];
  const int t = threadIdx.x;
  const int blockRow = blockIdx.x * 128;
  // stage x tile (f32 -> bf16), zero-pad rows >= N
  #pragma unroll 4
  for (int i = 0; i < 64; ++i) {
    int idx = t + i * 256;
    int row = idx >> 7, col = idx & 127;
    int gr = blockRow + row;
    float v = (gr < Nn) ? x[(size_t)gr * Hc + col] : 0.0f;
    xt[idx] = (__bf16)v;
  }
  __syncthreads();

  const int w = t >> 5, lane = t & 31;
  const int mrow = lane & 15, half = lane >> 4;
  const int rowBase = blockRow + w * 16;

  // A fragments (16x32 bf16): j=0..7 -> K = kt*32 + half*8 + j ; j=8..15 -> +16
  v16bf aF[4];
  #pragma unroll
  for (int kt = 0; kt < 4; ++kt) {
    const __bf16* p = &xt[(w * 16 + mrow) * Hc + kt * 32 + half * 8];
    v8bf lo = *(const v8bf*)p;
    v8bf hi = *(const v8bf*)(p + 16);
    #pragma unroll
    for (int j = 0; j < 8; ++j) { aF[kt][j] = lo[j]; aF[kt][j + 8] = hi[j]; }
  }

  #pragma unroll
  for (int nt = 0; nt < 8; ++nt) {
    v8f ai = {}, ac = {}, ao = {};
    #pragma unroll
    for (int kt = 0; kt < 4; ++kt) {
      const v16bf b0 = *(const v16bf*)(swz + (size_t)(((0 * 8 + nt) * 4 + kt) * 32 + lane) * 16);
      const v16bf b1 = *(const v16bf*)(swz + (size_t)(((1 * 8 + nt) * 4 + kt) * 32 + lane) * 16);
      const v16bf b2 = *(const v16bf*)(swz + (size_t)(((2 * 8 + nt) * 4 + kt) * 32 + lane) * 16);
      ai = wmma_bf16(aF[kt], b0, ai);
      ac = wmma_bf16(aF[kt], b1, ac);
      ao = wmma_bf16(aF[kt], b2, ao);
    }
    // C/D layout: col n = nt*16 + (lane&15); row = rowBase + r + 8*half
    const int n = nt * 16 + mrow;
    const float bi = bias[n], bc = bias[Hc + n], bo = bias[2 * Hc + n];
    const float wc2 = w_c[2 * Hc + n];
    #pragma unroll
    for (int r = 0; r < 8; ++r) {
      int row = rowBase + r + 8 * half;
      float I  = sigm(ai[r] + bi);
      float T  = tanhf(ac[r] + bc);
      float Cs = I * T;
      float O  = sigm(ao[r] + bo + wc2 * Cs);
      float Hv = O * tanhf(Cs);
      Hs[(size_t)row * Hc + n] = (__bf16)((row < Nn) ? Hv : 0.0f);
    }
  }
}

// ---------------------------------------------------------------------------
// GCN dense part: h = Hs @ gcn_w  (A read directly from bf16 Hs, 16B chunks)
// ---------------------------------------------------------------------------
__global__ __launch_bounds__(256)
void gcn_wmma_kernel(const __bf16* __restrict__ Hs, const __bf16* __restrict__ swz,
                     float* __restrict__ h) {
  const int t = threadIdx.x, w = t >> 5, lane = t & 31;
  const int mrow = lane & 15, half = lane >> 4;
  const int rowBase = blockIdx.x * 128 + w * 16;

  v16bf aF[4];
  #pragma unroll
  for (int kt = 0; kt < 4; ++kt) {
    const __bf16* p = Hs + (size_t)(rowBase + mrow) * Hc + kt * 32 + half * 8;
    v8bf lo = *(const v8bf*)p;
    v8bf hi = *(const v8bf*)(p + 16);
    #pragma unroll
    for (int j = 0; j < 8; ++j) { aF[kt][j] = lo[j]; aF[kt][j + 8] = hi[j]; }
  }

  #pragma unroll
  for (int nt = 0; nt < 8; ++nt) {
    v8f acc = {};
    #pragma unroll
    for (int kt = 0; kt < 4; ++kt) {
      const v16bf bf = *(const v16bf*)(swz + (size_t)(((3 * 8 + nt) * 4 + kt) * 32 + lane) * 16);
      acc = wmma_bf16(aF[kt], bf, acc);
    }
    const int n = nt * 16 + mrow;
    #pragma unroll
    for (int r = 0; r < 8; ++r) {
      int row = rowBase + r + 8 * half;
      h[(size_t)row * Hc + n] = acc[r];
    }
  }
}

// g = dg^2 * h  (self-loop term; also initializes the scatter accumulator)
__global__ void selfterm_kernel(const float* __restrict__ h, const float* __restrict__ dg,
                                float* __restrict__ g) {
  int idx = blockIdx.x * blockDim.x + threadIdx.x;    // NPAD*128 total
  int i = idx >> 7;
  float v = 0.0f;
  if (i < Nn) { float d = dg[i]; v = d * d * h[idx]; }
  g[idx] = v;
}

// g[dst] += dg[src]*dg[dst]*h[src] ; one thread per (edge, channel)
__global__ void scatter_kernel(const int* __restrict__ ei, const float* __restrict__ dg,
                               const float* __restrict__ h, float* __restrict__ g) {
  int idx = blockIdx.x * blockDim.x + threadIdx.x;
  int e = idx >> 7, c = idx & 127;
  if (e >= Ee) return;
  int s = ei[e], d = ei[Ee + e];
  float coef = dg[s] * dg[d];
  atomicAdd(&g[(size_t)d * Hc + c], coef * h[(size_t)s * Hc + c]);
}

// out = lin( BN( relu(g + gcn_b) ) ) ; one wave (32 lanes) per node, 4 ch/lane
__global__ __launch_bounds__(256)
void final_kernel(const float* __restrict__ g, const float* __restrict__ gcn_b,
                  const float* __restrict__ gamma, const float* __restrict__ beta,
                  const float* __restrict__ mean, const float* __restrict__ var,
                  const float* __restrict__ lin_w, const float* __restrict__ lin_b,
                  float* __restrict__ out) {
  int node = blockIdx.x * 8 + (threadIdx.x >> 5);
  int lane = threadIdx.x & 31;
  if (node >= Nn) return;
  float sum = 0.0f;
  #pragma unroll
  for (int q = 0; q < 4; ++q) {
    int c = lane * 4 + q;
    float v = g[(size_t)node * Hc + c] + gcn_b[c];
    v = fmaxf(v, 0.0f);
    v = gamma[c] * (v - mean[c]) * rsqrtf(var[c] + 1e-5f) + beta[c];
    sum += v * lin_w[c];
  }
  #pragma unroll
  for (int off = 16; off > 0; off >>= 1) sum += __shfl_down(sum, off, 32);
  if (lane == 0) out[node] = sum + lin_b[0];
}

extern "C" void kernel_launch(void* const* d_in, const int* in_sizes, int n_in,
                              void* d_out, int out_size, void* d_ws, size_t ws_size,
                              hipStream_t stream) {
  (void)in_sizes; (void)n_in; (void)out_size; (void)ws_size;
  const float* x       = (const float*)d_in[0];
  const int*   ei      = (const int*)d_in[1];
  /* d_in[2] edge_weight: dead (cheb collapses to bias)  */
  const float* W_x     = (const float*)d_in[3];
  const float* w_c     = (const float*)d_in[4];
  const float* b       = (const float*)d_in[5];
  /* d_in[6] theta: dead (multiplies zero state)          */
  const float* theta_b = (const float*)d_in[7];
  const float* gcn_w   = (const float*)d_in[8];
  const float* gcn_b   = (const float*)d_in[9];
  const float* bn_g    = (const float*)d_in[10];
  const float* bn_be   = (const float*)d_in[11];
  const float* bn_m    = (const float*)d_in[12];
  const float* bn_v    = (const float*)d_in[13];
  const float* lin_w   = (const float*)d_in[14];
  const float* lin_b   = (const float*)d_in[15];
  float* out = (float*)d_out;

  char* ws = (char*)d_ws;
  __bf16* swz  = (__bf16*)(ws + OFF_SWZ);
  float*  bias = (float*)(ws + OFF_BIAS);
  float*  deg  = (float*)(ws + OFF_DEG);
  float*  dg   = (float*)(ws + OFF_DG);
  __bf16* Hs   = (__bf16*)(ws + OFF_HS);
  float*  h    = (float*)(ws + OFF_H);
  float*  g    = (float*)(ws + OFF_G);

  prep_kernel<<<256, 256, 0, stream>>>(W_x, gcn_w, theta_b, b, swz, bias);
  deg_init_kernel<<<(NPAD + 255) / 256, 256, 0, stream>>>(deg);
  deg_count_kernel<<<(Ee + 255) / 256, 256, 0, stream>>>(ei, deg);
  dg_kernel<<<(NPAD + 255) / 256, 256, 0, stream>>>(deg, dg);
  gates_wmma_kernel<<<NPAD / 128, 256, 0, stream>>>(x, swz, bias, w_c, Hs);
  gcn_wmma_kernel<<<NPAD / 128, 256, 0, stream>>>(Hs, swz, h);
  selfterm_kernel<<<(NPAD * Hc) / 256, 256, 0, stream>>>(h, dg, g);
  scatter_kernel<<<(Ee * Hc) / 256, 256, 0, stream>>>(ei, dg, h, g);
  final_kernel<<<(Nn + 7) / 8, 256, 0, stream>>>(g, gcn_b, bn_g, bn_be, bn_m, bn_v,
                                                 lin_w, lin_b, out);
}
